// RGCN_65644280152931
// MI455X (gfx1250) — matrix operand order repaired and checked by reading, not compiled, not run on hardware
//
#include <hip/hip_runtime.h>

// ---------------------------------------------------------------------------
// RGCN (2 node types, 2 layers) for MI455X / gfx1250.
//  - Dense GEMMs use v_wmma_f32_16x16x32_bf16 (bf16 inputs, f32 accumulate),
//    32 output rows per wave; all B fragments for a k-step are preloaded so
//    the 8 WMMAs issue back-to-back behind a single s_wait_dscnt.
//  - SpMM uses float4 gathers + hardware global_atomic_add_f32.
// ---------------------------------------------------------------------------

typedef __attribute__((ext_vector_type(16))) __bf16 v16bf;
typedef __attribute__((ext_vector_type(8)))  float  v8f;

#define N_NODES 50000
#define F_IN    512
#define H_DIM   64
#define C_DIM   20
#define NPAD    50176          // 196 * 256 (256 rows per block, 32 per wave)
#define GEMM_BLOCKS (NPAD / 256)

// ---------------------------------------------------------------------------
// Zero-fill helper (grid-stride).
// ---------------------------------------------------------------------------
__global__ void zero_f32(float* __restrict__ p, size_t n) {
    size_t i = (size_t)blockIdx.x * blockDim.x + threadIdx.x;
    size_t stride = (size_t)gridDim.x * blockDim.x;
    for (; i < n; i += stride) p[i] = 0.0f;
}

// ---------------------------------------------------------------------------
// A-fragment loader: 16 bf16 per lane from a row-major f32 row.
// Lane layout (contiguous k order, matches B): run [k0..k0+15].
// ---------------------------------------------------------------------------
template <bool RELU>
__device__ __forceinline__ v16bf load_a_frag(const float* __restrict__ p, int k0) {
    float4 f0 = *(const float4*)(p + k0);
    float4 f1 = *(const float4*)(p + k0 + 4);
    float4 f2 = *(const float4*)(p + k0 + 8);
    float4 f3 = *(const float4*)(p + k0 + 12);
    if (RELU) {
        f0.x = fmaxf(f0.x, 0.f); f0.y = fmaxf(f0.y, 0.f);
        f0.z = fmaxf(f0.z, 0.f); f0.w = fmaxf(f0.w, 0.f);
        f1.x = fmaxf(f1.x, 0.f); f1.y = fmaxf(f1.y, 0.f);
        f1.z = fmaxf(f1.z, 0.f); f1.w = fmaxf(f1.w, 0.f);
        f2.x = fmaxf(f2.x, 0.f); f2.y = fmaxf(f2.y, 0.f);
        f2.z = fmaxf(f2.z, 0.f); f2.w = fmaxf(f2.w, 0.f);
        f3.x = fmaxf(f3.x, 0.f); f3.y = fmaxf(f3.y, 0.f);
        f3.z = fmaxf(f3.z, 0.f); f3.w = fmaxf(f3.w, 0.f);
    }
    v16bf a;
    a[0]  = (__bf16)f0.x; a[1]  = (__bf16)f0.y;
    a[2]  = (__bf16)f0.z; a[3]  = (__bf16)f0.w;
    a[4]  = (__bf16)f1.x; a[5]  = (__bf16)f1.y;
    a[6]  = (__bf16)f1.z; a[7]  = (__bf16)f1.w;
    a[8]  = (__bf16)f2.x; a[9]  = (__bf16)f2.y;
    a[10] = (__bf16)f2.z; a[11] = (__bf16)f2.w;
    a[12] = (__bf16)f3.x; a[13] = (__bf16)f3.y;
    a[14] = (__bf16)f3.z; a[15] = (__bf16)f3.w;
    return a;
}

// ---------------------------------------------------------------------------
// GEMM1: out[NPAD,64] = A[nrows,512] @ W[512,64]   (bf16 WMMA, f32 acc)
// Block: 256 threads = 8 waves; each wave owns a 32-row strip, all 4 N-tiles.
// Row pointers are CLAMPED (not guarded): padded rows of `out` are never
// consumed downstream (edge src indices < nrows), so garbage there is fine,
// and the inner loop stays straight-line (no exec divergence).
// W is staged to LDS transposed as bf16: WT[n][k]; each B fragment is one
// contiguous 32B LDS read per lane.
// ---------------------------------------------------------------------------
__global__ __launch_bounds__(256) void gemm_xw1(const float* __restrict__ A,
                                                const float* __restrict__ W,
                                                float* __restrict__ out,
                                                int nrows) {
    __shared__ __bf16 WT[H_DIM * F_IN];   // 64 KB, [n][k]
    const int tid = threadIdx.x;
    for (int idx = tid; idx < H_DIM * F_IN; idx += 256) {
        const int n = idx >> 9;           // idx / 512
        const int k = idx & 511;
        WT[idx] = (__bf16)W[k * H_DIM + n];
    }
    __syncthreads();

    const int wave  = tid >> 5;
    const int lane  = tid & 31;
    const int m     = lane & 15;
    const int khalf = lane >> 4;          // 0: lanes 0-15, 1: lanes 16-31
    const int row0  = blockIdx.x * 256 + wave * 32 + m;
    const int row1  = row0 + 16;
    const float* __restrict__ arow0 = A + (size_t)min(row0, nrows - 1) * F_IN;
    const float* __restrict__ arow1 = A + (size_t)min(row1, nrows - 1) * F_IN;

    v8f acc0[4] = {};
    v8f acc1[4] = {};

    for (int kstep = 0; kstep < F_IN / 32; ++kstep) {
        const int k0 = kstep * 32 + khalf * 16;
        const v16bf a0 = load_a_frag<false>(arow0, k0);
        const v16bf a1 = load_a_frag<false>(arow1, k0);
        // Preload all 4 B fragments (one LDS clause, one wait), then issue
        // the 8 WMMAs back-to-back.
        v16bf b[4];
        #pragma unroll
        for (int nt = 0; nt < 4; ++nt)
            b[nt] = *(const v16bf*)&WT[(nt * 16 + m) * F_IN +
                                       kstep * 32 + khalf * 16];
        #pragma unroll
        for (int nt = 0; nt < 4; ++nt) {
            acc0[nt] = __builtin_amdgcn_wmma_f32_16x16x32_bf16(
                false, a0, false, b[nt], (short)0, acc0[nt], false, false);
            acc1[nt] = __builtin_amdgcn_wmma_f32_16x16x32_bf16(
                false, a1, false, b[nt], (short)0, acc1[nt], false, false);
        }
    }

    // C/D layout: VGPR v -> row (v + khalf*8), col = lane&15
    const int rBase = blockIdx.x * 256 + wave * 32 + khalf * 8;
    #pragma unroll
    for (int nt = 0; nt < 4; ++nt) {
        #pragma unroll
        for (int v = 0; v < 8; ++v) {
            out[(size_t)(rBase + v) * H_DIM + nt * 16 + m]      = acc0[nt][v];
            out[(size_t)(rBase + 16 + v) * H_DIM + nt * 16 + m] = acc1[nt][v];
        }
    }
}

// ---------------------------------------------------------------------------
// GEMM2: out[NPAD,20] = relu(Hh[NPAD,64]) @ W[64,20]  (C padded to 32 in LDS)
// Hh is the zero-initialized padded hidden buffer, so all NPAD rows are
// readable and padded rows contribute zeros.
// ---------------------------------------------------------------------------
__global__ __launch_bounds__(256) void gemm_hw2(const float* __restrict__ Hh,
                                                const float* __restrict__ W,
                                                float* __restrict__ out) {
    __shared__ __bf16 WT[32 * H_DIM];     // 4 KB, [n][k], cols >= 20 zeroed
    const int tid = threadIdx.x;
    for (int idx = tid; idx < 32 * H_DIM; idx += 256) {
        const int n = idx >> 6;           // idx / 64
        const int k = idx & 63;
        WT[idx] = (__bf16)((n < C_DIM) ? W[k * C_DIM + n] : 0.0f);
    }
    __syncthreads();

    const int wave  = tid >> 5;
    const int lane  = tid & 31;
    const int m     = lane & 15;
    const int khalf = lane >> 4;
    const int row0  = blockIdx.x * 256 + wave * 32 + m;
    const float* __restrict__ arow0 = Hh + (size_t)row0 * H_DIM;
    const float* __restrict__ arow1 = arow0 + (size_t)16 * H_DIM;

    v8f acc0[2] = {};
    v8f acc1[2] = {};

    #pragma unroll
    for (int kstep = 0; kstep < H_DIM / 32; ++kstep) {
        const int k0 = kstep * 32 + khalf * 16;
        const v16bf a0 = load_a_frag<true>(arow0, k0);   // fused ReLU
        const v16bf a1 = load_a_frag<true>(arow1, k0);
        v16bf b[2];
        #pragma unroll
        for (int nt = 0; nt < 2; ++nt)
            b[nt] = *(const v16bf*)&WT[(nt * 16 + m) * H_DIM +
                                       kstep * 32 + khalf * 16];
        #pragma unroll
        for (int nt = 0; nt < 2; ++nt) {
            acc0[nt] = __builtin_amdgcn_wmma_f32_16x16x32_bf16(
                false, a0, false, b[nt], (short)0, acc0[nt], false, false);
            acc1[nt] = __builtin_amdgcn_wmma_f32_16x16x32_bf16(
                false, a1, false, b[nt], (short)0, acc1[nt], false, false);
        }
    }

    const int rBase = blockIdx.x * 256 + wave * 32 + khalf * 8;
    #pragma unroll
    for (int nt = 0; nt < 2; ++nt) {
        const int col = nt * 16 + m;
        if (col < C_DIM) {
            #pragma unroll
            for (int v = 0; v < 8; ++v) {
                out[(size_t)(rBase + v) * C_DIM + col]      = acc0[nt][v];
                out[(size_t)(rBase + 16 + v) * C_DIM + col] = acc1[nt][v];
            }
        }
    }
}

// ---------------------------------------------------------------------------
// SpMM scatter-add: y[dst] += vals[e] * x[src], row width LD floats,
// one thread per (edge, float4-chunk). LD*4 bytes is a multiple of 16 for
// both 64 and 20, so float4 accesses stay aligned.
// ---------------------------------------------------------------------------
template <int LD>
__global__ __launch_bounds__(256) void spmm_atomic(const int* __restrict__ eidx,
                                                   const float* __restrict__ vals,
                                                   const float* __restrict__ x,
                                                   float* __restrict__ y,
                                                   int nedges) {
    constexpr int DVEC = LD / 4;
    const int tid = blockIdx.x * 256 + threadIdx.x;
    const int e = tid / DVEC;
    const int c = tid % DVEC;
    if (e >= nedges) return;
    const int dst = eidx[e];
    const int src = eidx[nedges + e];
    const float v = vals[e];
    const float4 xx = *(const float4*)(x + (size_t)src * LD + c * 4);
    float* __restrict__ yp = y + (size_t)dst * LD + c * 4;
    unsafeAtomicAdd(yp + 0, v * xx.x);
    unsafeAtomicAdd(yp + 1, v * xx.y);
    unsafeAtomicAdd(yp + 2, v * xx.z);
    unsafeAtomicAdd(yp + 3, v * xx.w);
}

// ---------------------------------------------------------------------------
// Launcher.
// Workspace layout (floats):
//   xw[r]  = ws + r*NPAD*64              r = 0..3   (feat_j @ W1_r)
//   h[t]   = ws + (4+t)*NPAD*64          t = 0..1   (pre-relu hidden sums)
//   hw2[r] = ws + r*NPAD*20              (reuses xw region AFTER SpMM-1)
// Peak usage: 6*NPAD*64*4 B ~= 77 MB.
// ---------------------------------------------------------------------------
extern "C" void kernel_launch(void* const* d_in, const int* in_sizes, int n_in,
                              void* d_out, int out_size, void* d_ws, size_t ws_size,
                              hipStream_t stream) {
    const float* feat0 = (const float*)d_in[0];
    const float* feat1 = (const float*)d_in[1];
    const float* W1[4] = {(const float*)d_in[2], (const float*)d_in[3],
                          (const float*)d_in[4], (const float*)d_in[5]};
    const float* W2[4] = {(const float*)d_in[6], (const float*)d_in[7],
                          (const float*)d_in[8], (const float*)d_in[9]};
    const float* vals[4] = {(const float*)d_in[10], (const float*)d_in[11],
                            (const float*)d_in[12], (const float*)d_in[13]};
    const int* eidx[4] = {(const int*)d_in[14], (const int*)d_in[15],
                          (const int*)d_in[16], (const int*)d_in[17]};
    const int E = in_sizes[10];

    float* ws = (float*)d_ws;
    const size_t slab = (size_t)NPAD * H_DIM;
    float* xw[4]  = {ws, ws + slab, ws + 2 * slab, ws + 3 * slab};
    float* h[2]   = {ws + 4 * slab, ws + 5 * slab};
    float* hw2[4] = {ws, ws + (size_t)NPAD * C_DIM,
                     ws + 2 * (size_t)NPAD * C_DIM, ws + 3 * (size_t)NPAD * C_DIM};
    float* emb0 = (float*)d_out;
    float* emb1 = emb0 + (size_t)N_NODES * C_DIM;

    // 1. zero hidden accumulators (padded rows stay zero -> safe GEMM2 reads)
    zero_f32<<<4096, 256, 0, stream>>>(h[0], 2 * slab);

    // 2. layer-1 dense GEMMs (WMMA bf16)
    const float* srcFeat[4] = {feat0, feat1, feat0, feat1};
    for (int r = 0; r < 4; ++r)
        gemm_xw1<<<GEMM_BLOCKS, 256, 0, stream>>>(srcFeat[r], W1[r], xw[r], N_NODES);

    // 3. layer-1 SpMM scatter-add into h0 / h1
    const int spmm1Blocks = (E * (H_DIM / 4) + 255) / 256;
    for (int r = 0; r < 4; ++r)
        spmm_atomic<H_DIM><<<spmm1Blocks, 256, 0, stream>>>(
            eidx[r], vals[r], xw[r], (r < 2) ? h[0] : h[1], E);

    // 4. layer-2 dense GEMMs with fused ReLU (hw2 reuses the xw slabs;
    //    stream ordering guarantees SpMM-1 has finished reading them)
    const float* srcH[4] = {h[0], h[1], h[0], h[1]};
    for (int r = 0; r < 4; ++r)
        gemm_hw2<<<GEMM_BLOCKS, 256, 0, stream>>>(srcH[r], W2[r], hw2[r]);

    // 5. zero output, then layer-2 SpMM scatter-add into emb0 / emb1
    zero_f32<<<2048, 256, 0, stream>>>(emb0, (size_t)2 * N_NODES * C_DIM);
    const int spmm2Blocks = (E * (C_DIM / 4) + 255) / 256;
    for (int r = 0; r < 4; ++r)
        spmm_atomic<C_DIM><<<spmm2Blocks, 256, 0, stream>>>(
            eidx[r], vals[r], hw2[r], (r < 2) ? emb0 : emb1, E);
}
